// OEQTensorProduct_85744727097789
// MI455X (gfx1250) — matrix operand order repaired and checked by reading, not compiled, not run on hardware
//
#include <hip/hip_runtime.h>

typedef __attribute__((ext_vector_type(16))) __bf16 v16bf;
typedef __attribute__((ext_vector_type(8)))  float  v8f;

#define ETILE      16
#define SX_STRIDE  580           // padded float row (576+4) -> LDS bank skew
#define SY_STRIDE  12
#define LDS_SY_OFF 0
#define LDS_SX_OFF 768
#define LDS_SZ_OFF 37888         // 768 + 16*580*4
#define LDS_TOTAL  111872        // + (16*200 + 48*264 + 80*264)*2 bytes of Z

// ---------------- instruction metadata (from _build()) ----------------
// 11 paths: (i,j,k,local-in-k): k0 gets 3 paths (K=192), k1/k2 get 4 (K=256)
__constant__ unsigned char II[11]  = {0,0,0,1,1,1,1,2,2,2,2};
__constant__ unsigned char JJ[11]  = {0,1,2,0,1,1,2,0,1,2,2};
__constant__ unsigned char KK[11]  = {0,1,2,1,0,2,1,2,1,0,2};
__constant__ unsigned char LOC[11] = {0,0,0,1,1,1,2,2,3,2,3};
__constant__ int IOF1[3] = {0,64,256};
__constant__ int IOF2[3] = {0,1,4};
__constant__ int L1A[3]  = {1,3,5};
__constant__ int L3A[3]  = {1,3,5};
__constant__ int KD[3]   = {192,256,256};
__constant__ int ZSTRK[3]= {200,264,264};       // padded Z row stride (ushorts)
__constant__ int ZOFFK[3]= {0,3200,15872};      // ushort offsets of Z_k in LDS
__constant__ int AOFFK[3]= {0,12288,28672};     // ushort offsets of A_k in d_ws
__constant__ float ALPHA[3] = {0.07216878f, 0.10825318f, 0.13975425f};

// ---- sparse normalized real-Wigner-3j tables, sorted by (instr, r) ----
__constant__ unsigned char RS[11][6] = {
  {0,1,1,1,1,1},{1,2,3,4,4,4},{4,5,6,7,8,9},{9,10,11,12,12,12},
  {12,15,15,15,15,15},{15,17,19,22,24,26},{26,30,33,37,37,37},
  {37,38,39,40,41,42},{42,46,49,53,53,53},{53,58,58,58,58,58},
  {58,62,68,73,79,83}};
__constant__ unsigned char CP[83] = {
  0, 0,0,0, 0,0,0,0,0, 0,1,2, 0,1,2,
  2,0, 0,1, 1,0,2, 1,2, 2,0,
  2,1,0,0, 0,1,2, 0,2,1,2,
  0,1,2,3,4,
  0,1,2,4, 1,2,3, 0,2,3,4,
  0,1,2,3,4,
  1,3,0,2, 0,3,1,2,1,4, 0,1,3,2,4, 0,1,3,2,3,4, 1,3,2,4};
__constant__ unsigned char CQ[83] = {
  0, 0,1,2, 0,1,2,3,4, 0,0,0, 0,1,2,
  0,2, 1,0, 1,0,2, 2,1, 2,0,
  0,1,2,4, 1,2,3, 0,2,3,4,
  0,0,0,0,0,
  2,1,0,0, 0,1,2, 0,2,1,2,
  0,1,2,3,4,
  3,1,2,0, 3,0,2,1,4,1, 0,1,3,2,4, 1,0,2,3,4,3, 1,3,4,2};
__constant__ float CV[83] = {
  1.0f,
  0.57735027f,0.57735027f,0.57735027f,
  0.44721360f,0.44721360f,0.44721360f,0.44721360f,0.44721360f,
  0.57735027f,0.57735027f,0.57735027f,
  0.57735027f,0.57735027f,0.57735027f,
  0.31622777f,0.31622777f, 0.31622777f,0.31622777f,
  0.36514837f,-0.18257419f,-0.18257419f, 0.31622777f,0.31622777f,
  0.31622777f,-0.31622777f,
  0.31622777f,0.31622777f,-0.18257419f,-0.31622777f,
  0.31622777f,0.36514837f,0.31622777f,
  0.31622777f,-0.18257419f,0.31622777f,0.31622777f,
  0.44721360f,0.44721360f,0.44721360f,0.44721360f,0.44721360f,
  0.31622777f,0.31622777f,-0.18257419f,-0.31622777f,
  0.31622777f,0.36514837f,0.31622777f,
  0.31622777f,-0.18257419f,0.31622777f,0.31622777f,
  0.44721360f,0.44721360f,0.44721360f,0.44721360f,0.44721360f,
  0.20701967f,0.20701967f,-0.23904572f,-0.23904572f,
  0.20701967f,0.20701967f,0.11952286f,0.11952286f,-0.20701967f,-0.20701967f,
  -0.23904572f,0.11952286f,0.11952286f,0.23904572f,-0.23904572f,
  0.20701967f,0.20701967f,0.11952286f,0.11952286f,0.20701967f,0.20701967f,
  -0.20701967f,0.20701967f,-0.23904572f,-0.23904572f};

__device__ __forceinline__ unsigned short f2bf(float f) {
  unsigned u = __float_as_uint(f);
  u += 0x7fffu + ((u >> 16) & 1u);       // round-to-nearest-even
  return (unsigned short)(u >> 16);
}

// ---------------- prep: weights -> bf16 A_k = alpha * W^T, WMMA-A friendly ----
__global__ __launch_bounds__(256)
void tp_prep(const float* __restrict__ w, unsigned short* __restrict__ A) {
  int t = blockIdx.x * 256 + threadIdx.x;
  if (t >= 45056) return;
  int I  = t >> 12;               // weights flat layout == I*4096 + u*64 + ww
  int u  = (t >> 6) & 63;
  int ww = t & 63;
  int k  = KK[I];
  A[AOFFK[k] + ww * KD[k] + LOC[I] * 64 + u] = f2bf(w[t] * ALPHA[k]);
}

// ---------------- per-job GEMM tile: 16 w-rows x 16 edges, K-unrolled ----
template<int K, int NKB, int L3, int AOFFS, int ZOFFS, int ZSTR, int OUTOFF>
__device__ __forceinline__ void do_job(int wblk, int lane,
    const unsigned short* __restrict__ Aw, const unsigned short* sz, float* sout)
{
  const int m  = lane & 15;       // A-row (w) for A; edge column for B/D
  const int kg = lane >> 4;       // K-group select per documented layouts
  const unsigned short* Ab = Aw + AOFFS + (wblk * 16 + m) * K + kg * 8;
  v16bf af[NKB];
#pragma unroll
  for (int kb = 0; kb < NKB; ++kb) {          // A 16x32: K = kg*8 + {0..7,16..23}
    union { uint4 q[2]; v16bf v; } t;
    t.q[0] = *(const uint4*)(Ab + kb * 32);
    t.q[1] = *(const uint4*)(Ab + kb * 32 + 16);
    af[kb] = t.v;
  }
#pragma unroll
  for (int r = 0; r < L3; ++r) {
    v8f acc = {};
#pragma unroll
    for (int kb = 0; kb < NKB; ++kb) {        // B 32x16: K = kg*8 + {0..7,16..23}, n = m
      const unsigned short* bp = sz + ZOFFS + (r * 16 + m) * ZSTR + kb * 32 + kg * 8;
      union { uint4 q[2]; v16bf v; } t;
      t.q[0] = *(const uint4*)bp;
      t.q[1] = *(const uint4*)(bp + 16);
      acc = __builtin_amdgcn_wmma_f32_16x16x32_bf16(false, af[kb], false, t.v,
                                                    (short)0, acc, false, false);
    }
    // D 16x16 f32: lane n = m, rows M = kg*8 + v
    float* op = sout + m * SX_STRIDE + OUTOFF + r;
    const int w0 = wblk * 16 + kg * 8;
#pragma unroll
    for (int v = 0; v < 8; ++v) op[(w0 + v) * L3] = acc[v];
  }
}

// ---------------- main: 16 edges per 256-thread block ----------------
__global__ __launch_bounds__(256)
void tp_main(const float* __restrict__ x, const float* __restrict__ y,
             const unsigned short* __restrict__ Aw, float* __restrict__ out, int nE)
{
  extern __shared__ char lds[];
  float* sy = (float*)(lds + LDS_SY_OFF);
  float* sx = (float*)(lds + LDS_SX_OFF);
  unsigned short* sz = (unsigned short*)(lds + LDS_SZ_OFF);
  float* sout = sx;                              // reuse x staging after barrier

  const int tid = threadIdx.x;
  const int e0  = blockIdx.x * ETILE;

  // stage y
  for (int t = tid; t < ETILE * 9; t += 256) {
    int e = t / 9, c = t - e * 9;
    int ge = e0 + e; if (ge >= nE) ge = nE - 1;
    sy[e * SY_STRIDE + c] = y[(size_t)ge * 9 + c];
  }
  // stage x, fully coalesced b128
  const uint4* xg = (const uint4*)x;
  for (int t = tid; t < ETILE * 144; t += 256) {
    int e = t / 144, c = t - e * 144;
    int ge = e0 + e; if (ge >= nE) ge = nE - 1;
    ((uint4*)(sx + e * SX_STRIDE))[c] = xg[(size_t)ge * 144 + c];
  }
  // prefetch next block's x tile (global_prefetch_b8, speculative)
  if (e0 + ETILE < nE) {
    const char* base = (const char*)(x + (size_t)(e0 + ETILE) * 576);
    for (int t = tid; t < 288; t += 256) __builtin_prefetch(base + t * 128, 0, 1);
  }
  __syncthreads();

  // ---- phase Z: z_I[n,u,r] via sparse CG tables, bf16 into padded LDS ----
  for (int unit = tid; unit < 11 * 64 * ETILE; unit += 256) {
    int e = unit & 15;
    int u = (unit >> 4) & 63;
    int I = unit >> 10;
    const int i_ = II[I], j_ = JJ[I], k_ = KK[I];
    const int L1 = L1A[i_], L3 = L3A[k_];
    const float* xe = sx + e * SX_STRIDE + IOF1[i_] + u * L1;
    const float* ye = sy + e * SY_STRIDE + IOF2[j_];
    const int zstr = ZSTRK[k_];
    unsigned short* zp = sz + ZOFFK[k_] + LOC[I] * 64 + u;
    for (int r = 0; r < L3; ++r) {
      float a = 0.f;
      for (int c = RS[I][r]; c < RS[I][r + 1]; ++c)
        a += CV[c] * xe[CP[c]] * ye[CQ[c]];
      zp[(r * 16 + e) * zstr] = f2bf(a);
    }
  }
  __syncthreads();

  // ---- phase GEMM: 12 (k, w-block) jobs over 8 waves, balanced by cost ----
  const int wave = tid >> 5, lane = tid & 31;
  if (wave < 4) {          // k2 jobs: 5*8 = 40 wmma each
    do_job<256, 8, 5, 28672, 15872, 264, 256>(wave, lane, Aw, sz, sout);
  } else {                 // k1 (24 wmma) + k0 (6 wmma)
    do_job<256, 8, 3, 12288,  3200, 264,  64>(wave - 4, lane, Aw, sz, sout);
    do_job<192, 6, 1,     0,     0, 200,   0>(wave - 4, lane, Aw, sz, sout);
  }
  __syncthreads();

  // ---- coalesced b128 store of the 16x576 output tile ----
  uint4* og = (uint4*)out;
  for (int t = tid; t < ETILE * 144; t += 256) {
    int e = t / 144, c = t - e * 144;
    int ge = e0 + e;
    if (ge < nE) og[(size_t)ge * 144 + c] = ((const uint4*)(sout + e * SX_STRIDE))[c];
  }
}

extern "C" void kernel_launch(void* const* d_in, const int* in_sizes, int n_in,
                              void* d_out, int out_size, void* d_ws, size_t ws_size,
                              hipStream_t stream) {
  const float* x = (const float*)d_in[0];
  const float* y = (const float*)d_in[1];
  const float* w = (const float*)d_in[2];
  float* out = (float*)d_out;
  unsigned short* A = (unsigned short*)d_ws;    // 45056 bf16 = 90,112 B

  int nE = in_sizes[0] / 576;                   // 100000
  tp_prep<<<(45056 + 255) / 256, 256, 0, stream>>>(w, A);
  int blocks = (nE + ETILE - 1) / ETILE;        // 6250
  tp_main<<<blocks, 256, LDS_TOTAL, stream>>>(x, y, A, out, nE);
}
